// MemoryKoLeoLoss_72189810311269
// MI455X (gfx1250) — compile-verified, matrix operand order checked by reading, not run on hardware
//
#include <hip/hip_runtime.h>
#include <math.h>

// ---------------------------------------------------------------------------
// KoLeo loss, fused:  normalize -> (X Xt with row-max fused) -> -mean(log d)
// N = 8192 rows, D = 1024, f16 WMMA with fp32 accumulate on gfx1250 (wave32).
// Staging via Tensor Data Mover (tensor_load_to_lds) with double-buffered LDS
// when the builtin is available; falls back to global_load_b128 + ds_store.
// ---------------------------------------------------------------------------

typedef __attribute__((ext_vector_type(16))) _Float16 v16h;
typedef __attribute__((ext_vector_type(4)))  _Float16 v4h;
typedef __attribute__((ext_vector_type(8)))  float    v8f;
typedef __attribute__((ext_vector_type(4)))  float    f32x4;
typedef __attribute__((ext_vector_type(4)))  unsigned u32x4;

#define DIM      1024
#define BM       128
#define BN       128
#define BK       32
#define LDSB     40      // f16 per LDS row: 32 data + 8 pad = 80 bytes (20 banks)
#define NTHREADS 256     // 8 waves of 32

#if __has_builtin(__builtin_amdgcn_tensor_load_to_lds) && \
    __has_builtin(__builtin_amdgcn_s_wait_tensorcnt)
#define KOLEO_TDM 1
#else
#define KOLEO_TDM 0
#endif

union Frag {
    v16h  v;
    u32x4 q[2];
};

// monotone float -> uint key so we can use integer atomicMax for float max
__device__ __forceinline__ unsigned fkey(float f) {
    unsigned b = __float_as_uint(f);
    return (b & 0x80000000u) ? ~b : (b | 0x80000000u);
}
__device__ __forceinline__ float fdec(unsigned k) {
    unsigned b = (k & 0x80000000u) ? (k & 0x7FFFFFFFu) : ~k;
    return __uint_as_float(b);
}

#if KOLEO_TDM
typedef __attribute__((ext_vector_type(4))) unsigned tdm_v4u;
typedef __attribute__((ext_vector_type(8))) int      tdm_v8i;
typedef __attribute__((ext_vector_type(4))) int      tdm_v4i;

// LDS byte offset of a generic pointer into the shared segment
__device__ __forceinline__ unsigned lds_offset(const void* p) {
    return (unsigned)(uintptr_t)(__attribute__((address_space(3))) const void*)p;
}

// TDM: load a 2D tile of 128 rows x 32 f16 (64B/row) from a row-major f16
// tensor (row stride DIM) into LDS, padding +16B after every 64B row so the
// LDS layout has an 80B row stride (matches LDSB and is bank-conflict free).
// D# layout per CDNA5 ISA.
__device__ __forceinline__ void tdm_load_tile(const _Float16* gsrc,
                                              unsigned ldsByteOff, int nrows) {
    const unsigned long long ga = (unsigned long long)(uintptr_t)gsrc;

    tdm_v4u g0;
    g0[0] = 1u;                                          // count=1, user D#
    g0[1] = ldsByteOff;                                  // lds_addr
    g0[2] = (unsigned)(ga & 0xFFFFFFFFu);                // global_addr[31:0]
    g0[3] = (unsigned)((ga >> 32) & 0x01FFFFFFu)         // global_addr[56:32]
          | (2u << 30);                                  // type = 2 ("image")

    tdm_v8i g1;
    g1[0] = (1 << 16)        // data_size = 2 bytes
          | (1 << 20)        // pad_enable (LDS dest padding on load)
          | (3 << 22)        // pad_interval: code 3 = 16 DWORDs = 64B
          | (3 << 25);       // pad_amount:   code 3 =  4 DWORDs = 16B
    g1[1] = (int)((DIM & 0xFFFF) << 16);                 // tensor_dim0[15:0]
    g1[2] = (int)((DIM >> 16) | ((nrows & 0xFFFF) << 16)); // dim0 hi | dim1 lo
    g1[3] = (int)(((unsigned)nrows >> 16) | (BK << 16)); // dim1 hi | tile_dim0=32
    g1[4] = BM;                                          // tile_dim1=128, tile_dim2=0
    g1[5] = DIM;                                         // tensor_dim0_stride lo
    g1[6] = 0;                                           // stride hi | dim1_stride lo
    g1[7] = 0;                                           // dim1_stride hi (2D: unused)

    const tdm_v4i z4 = {};
#if defined(__clang_major__) && (__clang_major__ >= 23)
    const tdm_v8i z8 = {};
    __builtin_amdgcn_tensor_load_to_lds(g0, g1, z4, z4, z8, 0);
#else
    __builtin_amdgcn_tensor_load_to_lds(g0, g1, z4, z4, 0);
#endif
}
#endif // KOLEO_TDM

// Load A/B fragments from an LDS panel and run the 2x4 WMMA block.
__device__ __forceinline__ void compute_slab(const _Float16* __restrict__ AsP,
                                             const _Float16* __restrict__ BsP,
                                             int mWave, int nWave, int l15, int kOff,
                                             v8f acc[2][4]) {
    Frag fa[2], fb[4];
    #pragma unroll
    for (int f = 0; f < 2; ++f) {
        const _Float16* p = AsP + (mWave + f * 16 + l15) * LDSB + kOff;
        fa[f].q[0] = *(const u32x4*)(p);
        fa[f].q[1] = *(const u32x4*)(p + 16);
    }
    #pragma unroll
    for (int j = 0; j < 4; ++j) {
        const _Float16* p = BsP + (nWave + j * 16 + l15) * LDSB + kOff;
        fb[j].q[0] = *(const u32x4*)(p);
        fb[j].q[1] = *(const u32x4*)(p + 16);
    }
    #pragma unroll
    for (int f = 0; f < 2; ++f)
        #pragma unroll
        for (int j = 0; j < 4; ++j)
            acc[f][j] = __builtin_amdgcn_wmma_f32_16x16x32_f16(
                false, fa[f].v, false, fb[j].v, (short)0, acc[f][j], false, false);
}

// ---------------------------------------------------------------------------
// 1) L2-normalize each row, emit f16. One block per row; 256 thr * float4.
// ---------------------------------------------------------------------------
__global__ __launch_bounds__(NTHREADS)
void koleo_normalize(const float* __restrict__ X, _Float16* __restrict__ Xh) {
    const int row = blockIdx.x;
    const int t   = threadIdx.x;

    const f32x4 v = ((const f32x4*)(X + (size_t)row * DIM))[t];
    float ss = v[0]*v[0] + v[1]*v[1] + v[2]*v[2] + v[3]*v[3];

    #pragma unroll
    for (int o = 16; o > 0; o >>= 1) ss += __shfl_xor(ss, o, 32);

    __shared__ float red[NTHREADS / 32];
    if ((t & 31) == 0) red[t >> 5] = ss;
    __syncthreads();

    float tot = 0.f;
    #pragma unroll
    for (int i = 0; i < NTHREADS / 32; ++i) tot += red[i];

    const float scale = 1.0f / fmaxf(sqrtf(tot), 1e-12f);

    v4h h;
    h[0] = (_Float16)(v[0] * scale);
    h[1] = (_Float16)(v[1] * scale);
    h[2] = (_Float16)(v[2] * scale);
    h[3] = (_Float16)(v[3] * scale);
    ((v4h*)(Xh + (size_t)row * DIM))[t] = h;
}

// ---------------------------------------------------------------------------
// 2) init row-max keys (key 0 < key(-inf), so 0 is a safe identity)
// ---------------------------------------------------------------------------
__global__ void koleo_init(unsigned* __restrict__ rowMaxKey, int n) {
    int i = blockIdx.x * blockDim.x + threadIdx.x;
    if (i < n) rowMaxKey[i] = 0u;
}

// ---------------------------------------------------------------------------
// 3) Fused GEMM + row-max.  Block computes a 128x128 tile of X*Xt.
//    8 waves in a 4(M) x 2(N) grid; each wave: 2x4 wmma accumulators.
// ---------------------------------------------------------------------------
__global__ __launch_bounds__(NTHREADS)
void koleo_gemm_rowmax(const _Float16* __restrict__ Xh,
                       unsigned* __restrict__ rowMaxKey, int nrows) {
#if KOLEO_TDM
    __shared__ _Float16 As[2][BM * LDSB];   // double-buffered for TDM pipeline
    __shared__ _Float16 Bs[2][BN * LDSB];
#else
    __shared__ _Float16 As[1][BM * LDSB];
    __shared__ _Float16 Bs[1][BN * LDSB];
#endif
    __shared__ unsigned rowMaxLds[BM];

    const int tid  = threadIdx.x;
    const int wave = tid >> 5;
    const int lane = tid & 31;
    const int hi16 = lane >> 4;        // lane half selects K group
    const int l15  = lane & 15;
    const int kOff = hi16 * 8;         // f16 offset of first K chunk

    const int mWave = (wave & 3) * 32; // 4 waves along M
    const int nWave = (wave >> 2) * 64;// 2 waves along N

    const int rowBase = blockIdx.y * BM;
    const int colBase = blockIdx.x * BN;

    v8f acc[2][4] = {};

#if KOLEO_TDM
    // ---- TDM-staged, double-buffered pipeline --------------------------------
    const unsigned asOff0 = lds_offset(&As[0][0]);
    const unsigned asOff1 = lds_offset(&As[1][0]);
    const unsigned bsOff0 = lds_offset(&Bs[0][0]);
    const unsigned bsOff1 = lds_offset(&Bs[1][0]);

    if (wave == 0) {
        tdm_load_tile(Xh + (size_t)rowBase * DIM, asOff0, nrows);
        tdm_load_tile(Xh + (size_t)colBase * DIM, bsOff0, nrows);
    }

    int cur = 0;
    for (int kb = 0; kb < DIM; kb += BK, cur ^= 1) {
        __builtin_amdgcn_s_wait_tensorcnt(0);  // wave0: tile(kb) landed; others: nop
        __syncthreads();                       // tile visible; prev compute done
        if (wave == 0 && (kb + BK) < DIM) {    // prefetch tile(kb+BK) into other buffer
            tdm_load_tile(Xh + (size_t)rowBase * DIM + (kb + BK),
                          cur ? asOff0 : asOff1, nrows);
            tdm_load_tile(Xh + (size_t)colBase * DIM + (kb + BK),
                          cur ? bsOff0 : bsOff1, nrows);
        }
        compute_slab(&As[cur][0], &Bs[cur][0], mWave, nWave, l15, kOff, acc);
    }
#else
    // ---- fallback: per-lane global_load_b128 -> ds_store_b128 staging --------
    for (int kb = 0; kb < DIM; kb += BK) {
        #pragma unroll
        for (int i = 0; i < 2; ++i) {
            const int c = tid + i * NTHREADS;   // 0..511 chunk id
            const int r = c >> 2;               // panel row 0..127
            const int o = (c & 3) * 8;          // f16 offset 0/8/16/24
            const u32x4 av = *(const u32x4*)(Xh + (size_t)(rowBase + r) * DIM + kb + o);
            const u32x4 bv = *(const u32x4*)(Xh + (size_t)(colBase + r) * DIM + kb + o);
            *(u32x4*)(&As[0][r * LDSB + o]) = av;
            *(u32x4*)(&Bs[0][r * LDSB + o]) = bv;
        }
        __syncthreads();
        compute_slab(&As[0][0], &Bs[0][0], mWave, nWave, l15, kOff, acc);
        __syncthreads();
    }
#endif

    // ---- row-max epilogue: mask diagonal, reduce over the 128 columns ----
    if (tid < BM) rowMaxLds[tid] = 0u;
    __syncthreads();

    #pragma unroll
    for (int f = 0; f < 2; ++f) {
        #pragma unroll
        for (int e = 0; e < 8; ++e) {
            // C layout: VGPR e -> M = e (+8 for lanes 16-31), N = lane&15
            const int rowIn = mWave + f * 16 + e + hi16 * 8;
            const int rowG  = rowBase + rowIn;
            float mx = -3.0e38f;
            #pragma unroll
            for (int j = 0; j < 4; ++j) {
                const int colG = colBase + nWave + j * 16 + l15;
                const float vv = acc[f][j][e];
                mx = (rowG != colG) ? fmaxf(mx, vv) : mx;
            }
            atomicMax(&rowMaxLds[rowIn], fkey(mx));
        }
    }
    __syncthreads();

    if (tid < BM) atomicMax(&rowMaxKey[rowBase + tid], rowMaxLds[tid]);
}

// ---------------------------------------------------------------------------
// 4) final loss: -mean(log(sqrt(2 - 2*min(s,1) + 1e-4) + 1e-8))
// ---------------------------------------------------------------------------
__global__ __launch_bounds__(NTHREADS)
void koleo_loss(const unsigned* __restrict__ rowMaxKey, float* __restrict__ out, int n) {
    const int tid = threadIdx.x;
    float sum = 0.f;
    for (int i = tid; i < n; i += NTHREADS) {
        float s = fminf(fdec(rowMaxKey[i]), 1.0f);
        float d = sqrtf(fmaxf(2.0f - 2.0f * s + 1.0e-4f, 0.0f));
        sum += logf(d + 1.0e-8f);
    }
    #pragma unroll
    for (int o = 16; o > 0; o >>= 1) sum += __shfl_xor(sum, o, 32);

    __shared__ float red[NTHREADS / 32];
    if ((tid & 31) == 0) red[tid >> 5] = sum;
    __syncthreads();
    if (tid == 0) {
        float tot = 0.f;
        #pragma unroll
        for (int i = 0; i < NTHREADS / 32; ++i) tot += red[i];
        out[0] = -tot / (float)n;
    }
}

// ---------------------------------------------------------------------------
extern "C" void kernel_launch(void* const* d_in, const int* in_sizes, int n_in,
                              void* d_out, int out_size, void* d_ws, size_t ws_size,
                              hipStream_t stream) {
    const float* X = (const float*)d_in[0];
    const int N = in_sizes[0] / DIM;   // 8192

    _Float16* Xh        = (_Float16*)d_ws;                                  // 16 MB
    unsigned* rowMaxKey = (unsigned*)((char*)d_ws + (size_t)N * DIM * 2);   // 32 KB
    float*    out       = (float*)d_out;

    koleo_normalize<<<N, NTHREADS, 0, stream>>>(X, Xh);
    koleo_init<<<(N + NTHREADS - 1) / NTHREADS, NTHREADS, 0, stream>>>(rowMaxKey, N);

    dim3 grid(N / BN, N / BM);   // 64 x 64 = 4096 blocks
    koleo_gemm_rowmax<<<grid, NTHREADS, 0, stream>>>(Xh, rowMaxKey, N);

    koleo_loss<<<1, NTHREADS, 0, stream>>>(rowMaxKey, out, N);
}
